// ThreeOffsetsBlockShareWeights_87660282511813
// MI455X (gfx1250) — matrix operand order repaired and checked by reading, not compiled, not run on hardware
//
#include <hip/hip_runtime.h>
#include <hip/hip_bf16.h>
#include <stddef.h>

typedef __attribute__((ext_vector_type(16))) _Float16 v16h;
typedef __attribute__((ext_vector_type(4)))  _Float16 v4h;
typedef __attribute__((ext_vector_type(8)))  float    v8f;

#define SP 262144            // 64*64*64 spatial positions
#define CO 64                // output channels per conv
#define NB 2                 // batch

// per-tap offset directions (27 taps x 3 comps) from the reference
__device__ __constant__ float c_OFF[81] = {
  1,1,1,  1,1,0,  1,1,-1,  1,0,1,  1,0,0,  1,0,-1,  1,-1,1,  1,-1,0,  1,-1,-1,
  0,1,1,  0,1,0,  0,1,-1,  0,0,1,  0,0,0,  0,0,-1,  0,-1,1,  0,-1,0,  0,-1,-1,
 -1,1,1, -1,1,0, -1,1,-1, -1,0,1, -1,0,0, -1,0,-1, -1,-1,1, -1,-1,0, -1,-1,-1
};

// ---------------------------------------------------------------------------
// 1) Build effective 5x5x5 kernels eff[a][o][i][125] from W[o][i][27]
// ---------------------------------------------------------------------------
__global__ void build_eff(const float* __restrict__ W, float* __restrict__ eff,
                          int OI /* O*I */) {
  int idx = blockIdx.x * 256 + threadIdx.x;
  int total = 3 * OI * 125;
  if (idx >= total) return;
  int pos = idx % 125;
  int oi  = (idx / 125) % OI;
  int a   = idx / (125 * OI);
  int iz = pos / 25, iy = (pos / 5) % 5, ix = pos % 5;
  float alpha = (a == 0) ? 0.0f : ((a == 1) ? 0.4f : 0.7f);
  const float* Wk = W + (size_t)oi * 27;
  float acc = 0.0f;
  for (int k = 0; k < 27; ++k) {
    float pz = (float)(k / 9) - 1.0f;
    float py = (float)((k / 3) % 3) - 1.0f;
    float px = (float)(k % 3) - 1.0f;
    float dz = pz + alpha * c_OFF[k * 3 + 0];
    float dy = py + alpha * c_OFF[k * 3 + 1];
    float dx = px + alpha * c_OFF[k * 3 + 2];
    float fz = floorf(dz), fy = floorf(dy), fx = floorf(dx);
    float tz = dz - fz, ty = dy - fy, tx = dx - fx;
    int bz = (int)fz + 2, by = (int)fy + 2, bx = (int)fx + 2;
    float wv = Wk[k];
    #pragma unroll
    for (int cz = 0; cz < 2; ++cz)
      #pragma unroll
      for (int cy = 0; cy < 2; ++cy)
        #pragma unroll
        for (int cx = 0; cx < 2; ++cx) {
          float wt = (cz ? tz : 1.0f - tz) * (cy ? ty : 1.0f - ty) * (cx ? tx : 1.0f - tx);
          if (wt != 0.0f && (bz + cz) == iz && (by + cy) == iy && (bx + cx) == ix)
            acc += wt * wv;
        }
  }
  eff[idx] = acc;   // layout ((a*OI)+oi)*125 + pos
}

// ---------------------------------------------------------------------------
// 2) Pack eff into the v_wmma_f32_16x16x32_f16 A-fragment layout.
//    effA[(((a*4+mt)*KC + kc)*32 + lane)*16 + j]; K order = tap-major, ch-minor.
//    A layout used here (contiguous k, same order as the B reads): M = lane&15; base = lane<16 ? 0 : 16;
//    K = base+j.
// ---------------------------------------------------------------------------
__global__ void pack_effA(const float* __restrict__ eff, _Float16* __restrict__ effA,
                          int O, int I, int KC) {
  int idx = blockIdx.x * 256 + threadIdx.x;
  int total = 3 * 4 * KC * 512;
  if (idx >= total) return;
  int j    = idx & 15;
  int lane = (idx >> 4) & 31;
  int rest = idx >> 9;
  int kc = rest % KC;
  int mt = (rest / KC) & 3;
  int a  = rest / (KC * 4);
  int M = lane & 15;
  int base = (lane < 16) ? 0 : 16;
  int K = base + j;
  int kgl = kc * 32 + K;
  int tap = kgl / I;
  int ch  = kgl % I;
  int o = mt * 16 + M;
  effA[idx] = (_Float16)eff[(((size_t)a * O + o) * I + ch) * 125 + tap];
}

// Pack Wc[64][192] into A fragments: WcA[((mt*6+kc)*32+lane)*16+j]
__global__ void pack_wc(const float* __restrict__ Wc, _Float16* __restrict__ WcA) {
  int idx = blockIdx.x * 256 + threadIdx.x;
  if (idx >= 4 * 6 * 512) return;
  int j = idx & 15, lane = (idx >> 4) & 31;
  int rest = idx >> 9;
  int kc = rest % 6, mt = rest / 6;
  int M = lane & 15;
  int base = (lane < 16) ? 0 : 16;
  int K = base + j;
  WcA[idx] = (_Float16)Wc[(size_t)(mt * 16 + M) * 192 + kc * 32 + K];
}

// ---------------------------------------------------------------------------
// 3) Implicit-GEMM 5x5x5 conv via WMMA.
//    Block = 128 threads (4 waves). Wave = one 16-row M-tile (out channels);
//    block N-tile = full 64-wide W row -> 4 N-subtiles / 4 accumulators.
//    Double-buffered LDS row tile [w+2][c]: stage tile it+1 while computing
//    tile it; one barrier per iteration. Loader: 4 channels x 4 w per thread
//    -> global_load_b128 reads + packed ds_store_b64 writes.
//    B fragment: lane l, elem j = ldsT[p][(l&15)+n0+dx][16*(l>>4)+j]  -> one
//    aligned 32B LDS read at a compile-time immediate offset.
//    z layout: [((a*2+b)*64 + o)][SP]
// ---------------------------------------------------------------------------
template <int S>   // K-subchunks per tap; CIN = 32*S
__global__ __launch_bounds__(128)
void conv_wmma(const float* __restrict__ x, const _Float16* __restrict__ effA,
               const float* __restrict__ bias, float* __restrict__ z) {
  constexpr int CIN = 32 * S;
  constexpr int KC  = 125 * S;
  constexpr int NIT = 25 * S;
  __shared__ __align__(32) _Float16 ldsT[2][68][32];   // [buf][w+2][c]
  int bid = blockIdx.x;                // 3 * 2 * 64 * 64 blocks
  int h = bid & 63;
  int d = (bid >> 6) & 63;
  int b = (bid >> 12) & 1;
  int a = bid >> 13;
  int tid = threadIdx.x;
  int lane = tid & 31;
  int wv = tid >> 5;                   // wave id == M-tile
  int c0  = (tid & 7) * 4;             // loader channel base 0,4,..,28
  int w0g = (tid >> 3) * 4;            // loader w base 0,4,..,60
  // zero halo rows 0,1,66,67 in both buffers (never rewritten)
  {
    int hw = tid >> 5;                               // 0..3
    int hrow = (hw < 2) ? hw : (64 + hw);            // 0,1,66,67
    ldsT[0][hrow][tid & 31] = (_Float16)0.0f;
    ldsT[1][hrow][tid & 31] = (_Float16)0.0f;
  }
  const float* xb = x + (size_t)b * CIN * SP;
  const _Float16* Abase = effA + (size_t)(a * 4 + wv) * KC * 512;
  v8f acc0 = {}, acc1 = {}, acc2 = {}, acc3 = {};

  auto stage = [&](int it, int p) {
    int g = it / S, sub = it - g * S;
    int dz = g / 5, dy = g - dz * 5;
    int zz = d + dz - 2, yy = h + dy - 2;
    bool rowok = (zz >= 0) && (zz < 64) && (yy >= 0) && (yy < 64);
    long rowoff = ((long)zz * 64 + yy) * 64;
    const float* xr = xb + (size_t)(sub * 32 + c0) * SP + rowoff + w0g;
    __builtin_prefetch(xr + 64, 0, 1);   // next dy row -> global_prefetch
    float fv[4][4];
    #pragma unroll
    for (int ch = 0; ch < 4; ++ch) {
      float4 t = rowok ? *(const float4*)(xr + (size_t)ch * SP)
                       : make_float4(0.0f, 0.0f, 0.0f, 0.0f);
      fv[ch][0] = t.x; fv[ch][1] = t.y; fv[ch][2] = t.z; fv[ch][3] = t.w;
    }
    #pragma unroll
    for (int q = 0; q < 4; ++q) {
      v4h pk = { (_Float16)fv[0][q], (_Float16)fv[1][q],
                 (_Float16)fv[2][q], (_Float16)fv[3][q] };
      *(v4h*)(&ldsT[p][2 + w0g + q][c0]) = pk;       // one ds_store_b64
    }
  };

  stage(0, 0);
  __syncthreads();
  for (int it = 0; it < NIT; ++it) {
    int p = it & 1;
    if (it + 1 < NIT) stage(it + 1, p ^ 1);          // overlap with WMMAs
    int g = it / S, sub = it - g * S;
    int kc0 = g * 5 * S + sub;
    const _Float16* fragBase = &ldsT[p][lane & 15][(lane >> 4) << 4];
    #pragma unroll
    for (int dx = 0; dx < 5; ++dx) {
      v16h afrag = *(const v16h*)(Abase + ((size_t)(kc0 + dx * S) * 32 + lane) * 16);
      v16h b0 = *(const v16h*)(fragBase + (dx +  0) * 32);
      v16h b1 = *(const v16h*)(fragBase + (dx + 16) * 32);
      v16h b2 = *(const v16h*)(fragBase + (dx + 32) * 32);
      v16h b3 = *(const v16h*)(fragBase + (dx + 48) * 32);
      acc0 = __builtin_amdgcn_wmma_f32_16x16x32_f16(false, afrag, false, b0, (short)0, acc0, false, false);
      acc1 = __builtin_amdgcn_wmma_f32_16x16x32_f16(false, afrag, false, b1, (short)0, acc1, false, false);
      acc2 = __builtin_amdgcn_wmma_f32_16x16x32_f16(false, afrag, false, b2, (short)0, acc2, false, false);
      acc3 = __builtin_amdgcn_wmma_f32_16x16x32_f16(false, afrag, false, b3, (short)0, acc3, false, false);
    }
    __syncthreads();
  }
  // C/D layout: VGPR r -> M = r + 8*(lane>=16), N = lane&15
  int n  = lane & 15;
  int hi = (lane >= 16) ? 8 : 0;
  size_t posbase = ((size_t)d * 64 + h) * 64 + n;
  #pragma unroll
  for (int r = 0; r < 8; ++r) {
    int o = wv * 16 + r + hi;
    float bv = bias[o];
    float* row = z + ((size_t)((a * 2 + b) * CO + o)) * SP + posbase;
    row[0]  = acc0[r] + bv;
    row[16] = acc1[r] + bv;
    row[32] = acc2[r] + bv;
    row[48] = acc3[r] + bv;
  }
}

// ---------------------------------------------------------------------------
// 4) BN stats (training mode, biased) -> fused scale/shift. float4 streaming.
// ---------------------------------------------------------------------------
__global__ __launch_bounds__(256)
void stats_cat(const float* __restrict__ z, const float* __restrict__ g,
               const float* __restrict__ bt, float* __restrict__ scale,
               float* __restrict__ shift) {
  int c = blockIdx.x;            // 0..191 concat channel
  int a = c / CO, o = c % CO;
  __shared__ float ss[256], sq[256];
  float s = 0.0f, q2 = 0.0f;
  for (int b = 0; b < NB; ++b) {
    const float4* p = (const float4*)(z + ((size_t)((a * 2 + b) * CO + o)) * SP);
    for (int i = threadIdx.x; i < SP / 4; i += 256) {
      float4 v = p[i];
      s  += v.x + v.y + v.z + v.w;
      q2 += v.x * v.x + v.y * v.y + v.z * v.z + v.w * v.w;
    }
  }
  ss[threadIdx.x] = s; sq[threadIdx.x] = q2;
  __syncthreads();
  for (int st = 128; st > 0; st >>= 1) {
    if (threadIdx.x < st) { ss[threadIdx.x] += ss[threadIdx.x + st]; sq[threadIdx.x] += sq[threadIdx.x + st]; }
    __syncthreads();
  }
  if (threadIdx.x == 0) {
    float N = (float)(NB * SP);
    float mu = ss[0] / N;
    float var = sq[0] / N - mu * mu;
    float rs = rsqrtf(var + 1e-5f);
    float sc = rs * g[o];
    scale[c] = sc;
    shift[c] = bt[o] - mu * sc;
  }
}

__global__ __launch_bounds__(256)
void stats_mix(const float* __restrict__ u, const float* __restrict__ g,
               const float* __restrict__ bt, float* __restrict__ scale,
               float* __restrict__ shift) {
  int o = blockIdx.x;            // 0..63
  __shared__ float ss[256], sq[256];
  float s = 0.0f, q2 = 0.0f;
  for (int b = 0; b < NB; ++b) {
    const float4* p = (const float4*)(u + ((size_t)(b * CO + o)) * SP);
    for (int i = threadIdx.x; i < SP / 4; i += 256) {
      float4 v = p[i];
      s  += v.x + v.y + v.z + v.w;
      q2 += v.x * v.x + v.y * v.y + v.z * v.z + v.w * v.w;
    }
  }
  ss[threadIdx.x] = s; sq[threadIdx.x] = q2;
  __syncthreads();
  for (int st = 128; st > 0; st >>= 1) {
    if (threadIdx.x < st) { ss[threadIdx.x] += ss[threadIdx.x + st]; sq[threadIdx.x] += sq[threadIdx.x + st]; }
    __syncthreads();
  }
  if (threadIdx.x == 0) {
    float N = (float)(NB * SP);
    float mu = ss[0] / N;
    float var = sq[0] / N - mu * mu;
    float rs = rsqrtf(var + 1e-5f);
    float sc = rs * g[o];
    scale[o] = sc;
    shift[o] = bt[o] - mu * sc;
  }
}

// ---------------------------------------------------------------------------
// 5) Channel mix (192->64 GEMM), BN+ReLU fused into B-tile build. N-tile = 64.
//    u layout: [(b*64 + o)][SP]
// ---------------------------------------------------------------------------
__global__ __launch_bounds__(128)
void mix_wmma(const float* __restrict__ z, const _Float16* __restrict__ WcA,
              const float* __restrict__ scale, const float* __restrict__ shift,
              const float* __restrict__ bc, float* __restrict__ u) {
  __shared__ __align__(32) _Float16 ldsT[64][32];   // [n][c]
  int bid = blockIdx.x;                // 2 * 64 * 64 blocks
  int h = bid & 63;
  int d = (bid >> 6) & 63;
  int b = (bid >> 12) & 1;
  int tid = threadIdx.x;
  int lane = tid & 31;
  int wv = tid >> 5;
  size_t posbase = ((size_t)d * 64 + h) * 64;
  int c0  = (tid & 7) * 4;
  int w0g = (tid >> 3) * 4;
  const _Float16* fragBase = &ldsT[lane & 15][(lane >> 4) << 4];
  v8f acc0 = {}, acc1 = {}, acc2 = {}, acc3 = {};
  for (int kc = 0; kc < 6; ++kc) {
    float fv[4][4];
    #pragma unroll
    for (int ch = 0; ch < 4; ++ch) {
      int ccat = kc * 32 + c0 + ch;
      int aa = ccat >> 6, o = ccat & 63;
      const float4* p = (const float4*)(z + ((size_t)((aa * 2 + b) * CO + o)) * SP + posbase + w0g);
      float sc = scale[ccat], sh = shift[ccat];
      float4 t = *p;
      fv[ch][0] = fmaxf(t.x * sc + sh, 0.0f);
      fv[ch][1] = fmaxf(t.y * sc + sh, 0.0f);
      fv[ch][2] = fmaxf(t.z * sc + sh, 0.0f);
      fv[ch][3] = fmaxf(t.w * sc + sh, 0.0f);
    }
    __syncthreads();
    #pragma unroll
    for (int q = 0; q < 4; ++q) {
      v4h pk = { (_Float16)fv[0][q], (_Float16)fv[1][q],
                 (_Float16)fv[2][q], (_Float16)fv[3][q] };
      *(v4h*)(&ldsT[w0g + q][c0]) = pk;
    }
    __syncthreads();
    v16h afrag = *(const v16h*)(WcA + ((size_t)(wv * 6 + kc) * 32 + lane) * 16);
    v16h b0 = *(const v16h*)(fragBase +  0 * 32);
    v16h b1 = *(const v16h*)(fragBase + 16 * 32);
    v16h b2 = *(const v16h*)(fragBase + 32 * 32);
    v16h b3 = *(const v16h*)(fragBase + 48 * 32);
    acc0 = __builtin_amdgcn_wmma_f32_16x16x32_f16(false, afrag, false, b0, (short)0, acc0, false, false);
    acc1 = __builtin_amdgcn_wmma_f32_16x16x32_f16(false, afrag, false, b1, (short)0, acc1, false, false);
    acc2 = __builtin_amdgcn_wmma_f32_16x16x32_f16(false, afrag, false, b2, (short)0, acc2, false, false);
    acc3 = __builtin_amdgcn_wmma_f32_16x16x32_f16(false, afrag, false, b3, (short)0, acc3, false, false);
  }
  int n = lane & 15;
  int hi = (lane >= 16) ? 8 : 0;
  #pragma unroll
  for (int r = 0; r < 8; ++r) {
    int o = wv * 16 + r + hi;
    float bv = bc[o];
    float* row = u + ((size_t)(b * CO + o)) * SP + posbase + n;
    row[0]  = acc0[r] + bv;
    row[16] = acc1[r] + bv;
    row[32] = acc2[r] + bv;
    row[48] = acc3[r] + bv;
  }
}

// 6) Elementwise BN+ReLU, float4. total = NB*CO*SP/4 lanes exactly; SP/4 = 65536.
__global__ void bnrelu(const float4* __restrict__ u, const float* __restrict__ sc,
                       const float* __restrict__ sh, float4* __restrict__ out) {
  unsigned idx = blockIdx.x * 256u + threadIdx.x;
  int ch = (int)((idx >> 16) & (CO - 1));
  float s = sc[ch], t = sh[ch];
  float4 v = u[idx];
  float4 r;
  r.x = fmaxf(v.x * s + t, 0.0f);
  r.y = fmaxf(v.y * s + t, 0.0f);
  r.z = fmaxf(v.z * s + t, 0.0f);
  r.w = fmaxf(v.w * s + t, 0.0f);
  out[idx] = r;
}

// ---------------------------------------------------------------------------
extern "C" void kernel_launch(void* const* d_in, const int* in_sizes, int n_in,
                              void* d_out, int out_size, void* d_ws, size_t ws_size,
                              hipStream_t stream) {
  const float* x   = (const float*)d_in[0];
  const float* W1  = (const float*)d_in[1];
  const float* b1  = (const float*)d_in[2];
  const float* g1  = (const float*)d_in[3];
  const float* bt1 = (const float*)d_in[4];
  const float* W2  = (const float*)d_in[5];
  const float* b2  = (const float*)d_in[6];
  const float* g2  = (const float*)d_in[7];
  const float* bt2 = (const float*)d_in[8];
  const float* Wc1 = (const float*)d_in[9];
  const float* bc1 = (const float*)d_in[10];
  const float* gc1 = (const float*)d_in[11];
  const float* btc1= (const float*)d_in[12];
  const float* Wc2 = (const float*)d_in[13];
  const float* bc2 = (const float*)d_in[14];
  const float* gc2 = (const float*)d_in[15];
  const float* btc2= (const float*)d_in[16];
  float* out = (float*)d_out;

  // carve workspace
  char* ws = (char*)d_ws;
  size_t off = 0;
  auto carve = [&](size_t bytes) { size_t o = off; off = (off + bytes + 255) & ~(size_t)255; return o; };
  float*    eff1   = (float*)   (ws + carve((size_t)3*64*32*125*4));
  float*    eff2   = (float*)   (ws + carve((size_t)3*64*64*125*4));
  _Float16* effA1  = (_Float16*)(ws + carve((size_t)3*4*125*512*2));
  _Float16* effA2  = (_Float16*)(ws + carve((size_t)3*4*250*512*2));
  _Float16* wcA1   = (_Float16*)(ws + carve((size_t)4*6*512*2));
  _Float16* wcA2   = (_Float16*)(ws + carve((size_t)4*6*512*2));
  float*    sc_cat = (float*)   (ws + carve(192*4));
  float*    sh_cat = (float*)   (ws + carve(192*4));
  float*    sc_mx  = (float*)   (ws + carve(64*4));
  float*    sh_mx  = (float*)   (ws + carve(64*4));
  float*    zbuf   = (float*)   (ws + carve((size_t)3*NB*CO*SP*4));  // 403 MB
  float*    ubuf   = (float*)   (ws + carve((size_t)NB*CO*SP*4));    // 134 MB
  float*    xs     = (float*)   (ws + carve((size_t)NB*CO*SP*4));    // 134 MB

  const int convGrid = 3 * NB * 64 * 64;   // 24576 (alpha, batch, d, h)
  const int mixGrid  = NB * 64 * 64;       // 8192
  const int ewGrid   = (NB * CO * SP / 4) / 256;  // 32768

  // ---- prep: effective kernels + WMMA A-fragment packing ----
  build_eff<<<3000, 256, 0, stream>>>(W1, eff1, 64 * 32);
  build_eff<<<6000, 256, 0, stream>>>(W2, eff2, 64 * 64);
  pack_effA<<<3000, 256, 0, stream>>>(eff1, effA1, 64, 32, 125);
  pack_effA<<<6000, 256, 0, stream>>>(eff2, effA2, 64, 64, 250);
  pack_wc<<<48, 256, 0, stream>>>(Wc1, wcA1);
  pack_wc<<<48, 256, 0, stream>>>(Wc2, wcA2);

  // ---- stage 1 ----
  conv_wmma<1><<<convGrid, 128, 0, stream>>>(x, effA1, b1, zbuf);
  stats_cat<<<192, 256, 0, stream>>>(zbuf, g1, bt1, sc_cat, sh_cat);
  mix_wmma<<<mixGrid, 128, 0, stream>>>(zbuf, wcA1, sc_cat, sh_cat, bc1, ubuf);
  stats_mix<<<64, 256, 0, stream>>>(ubuf, gc1, btc1, sc_mx, sh_mx);
  bnrelu<<<ewGrid, 256, 0, stream>>>((const float4*)ubuf, sc_mx, sh_mx, (float4*)xs);

  // ---- stage 2 ----
  conv_wmma<2><<<convGrid, 128, 0, stream>>>(xs, effA2, b2, zbuf);
  stats_cat<<<192, 256, 0, stream>>>(zbuf, g2, bt2, sc_cat, sh_cat);
  mix_wmma<<<mixGrid, 128, 0, stream>>>(zbuf, wcA2, sc_cat, sh_cat, bc2, ubuf);
  stats_mix<<<64, 256, 0, stream>>>(ubuf, gc2, btc2, sc_mx, sh_mx);
  bnrelu<<<ewGrid, 256, 0, stream>>>((const float4*)ubuf, sc_mx, sh_mx, (float4*)out);

  (void)in_sizes; (void)n_in; (void)out_size; (void)ws_size;
}
